// SAGE_38113539785173
// MI455X (gfx1250) — matrix operand order repaired and checked by reading, not compiled, not run on hardware
//
#include <hip/hip_runtime.h>

typedef __attribute__((ext_vector_type(16))) _Float16 v16h;
typedef __attribute__((ext_vector_type(8)))  float    v8f;

#define EPS_BN 1e-5f

// ---------------- utility kernels ----------------

__global__ void zero_f32(float* p, long long n) {
  long long i = (long long)blockIdx.x * blockDim.x + threadIdx.x;
  long long T = (long long)gridDim.x * blockDim.x;
  for (; i < n; i += T) p[i] = 0.0f;
}

__global__ void degree_kernel(const long long* dst, float* deg, int E) {
  int e = blockIdx.x * blockDim.x + threadIdx.x;
  if (e >= E) return;
  atomicAdd(&deg[(int)dst[e]], 1.0f);
}

__global__ void deg_inv_kernel(float* deg, int n) {
  int i = blockIdx.x * blockDim.x + threadIdx.x;
  if (i >= n) return;
  deg[i] = 1.0f / fmaxf(deg[i], 1.0f);
}

// one edge per 64 lanes: coalesced 256B row gather + f32 atomic scatter (L2-resident)
__global__ void scatter_kernel(const long long* __restrict__ src,
                               const long long* __restrict__ dst,
                               const float* __restrict__ feat,
                               float* __restrict__ agg, int E) {
  long long idx = (long long)blockIdx.x * blockDim.x + threadIdx.x;
  int e = (int)(idx >> 6);
  int c = (int)(idx & 63);
  if (e >= E) return;
  int s = (int)src[e];
  int d = (int)dst[e];
  atomicAdd(&agg[(long long)d * 64 + c], feat[(long long)s * 64 + c]);
}

// pack f32 weight [64 x outc] into wave32 WMMA B-fragments:
// flat layout [ntile][kchunk][lane][16 halves]; lanes 0-15: N=lane, K=kc*32+{0..7,16..23}
// lanes 16-31: N=lane-16, K=kc*32+{8..15,24..31}  (zero-pad columns >= outc)
__global__ void packB_kernel(const float* __restrict__ W, _Float16* __restrict__ out,
                             int outc, int ntiles) {
  int t = blockIdx.x * blockDim.x + threadIdx.x;
  int total = ntiles * 2 * 32 * 16;
  if (t >= total) return;
  int i    = t & 15;
  int lane = (t >> 4) & 31;
  int kc   = (t >> 9) & 1;
  int nt   = t >> 10;
  int n = nt * 16 + (lane & 15);
  int k = kc * 32 + (lane >> 4) * 8 + (i & 7) + (i >> 3) * 16;
  float v = (n < outc) ? W[k * outc + n] : 0.0f;
  out[t] = (_Float16)v;
}

// A-fragment for v_wmma_f32_16x16x32_f16 (16-bit A 16x32 layout):
// lane L: row M = m0 + (L&15); elems 0..7 = K[k0 + (L>>4)*8 .. +7], elems 8..15 = K[+16]
__device__ inline v16h load_a_frag(const float* __restrict__ row, int off, float s) {
  float4 a0 = *(const float4*)(row + off);
  float4 a1 = *(const float4*)(row + off + 4);
  float4 a2 = *(const float4*)(row + off + 16);
  float4 a3 = *(const float4*)(row + off + 20);
  v16h a;
  a[0]  = (_Float16)(a0.x * s); a[1]  = (_Float16)(a0.y * s);
  a[2]  = (_Float16)(a0.z * s); a[3]  = (_Float16)(a0.w * s);
  a[4]  = (_Float16)(a1.x * s); a[5]  = (_Float16)(a1.y * s);
  a[6]  = (_Float16)(a1.z * s); a[7]  = (_Float16)(a1.w * s);
  a[8]  = (_Float16)(a2.x * s); a[9]  = (_Float16)(a2.y * s);
  a[10] = (_Float16)(a2.z * s); a[11] = (_Float16)(a2.w * s);
  a[12] = (_Float16)(a3.x * s); a[13] = (_Float16)(a3.y * s);
  a[14] = (_Float16)(a3.z * s); a[15] = (_Float16)(a3.w * s);
  return a;
}

// out[m0:m0+16, :] = (agg*deginv) @ Wl + x @ Wr + bias
// one wave produces the full 16 x (NT*16) output strip: each A-fragment is reused
// across NT column tiles -> 2*2*NT chained WMMAs per wave, 4x less A-gather traffic.
template <int NT>
__global__ void sage_gemm_kernel(const float* __restrict__ agg,
                                 const float* __restrict__ deginv,
                                 const float* __restrict__ x,
                                 const _Float16* __restrict__ WlP,
                                 const _Float16* __restrict__ WrP,
                                 const float* __restrict__ bias,
                                 float* __restrict__ out,
                                 int nodes, int outc) {
  int wave = blockIdx.x * (blockDim.x >> 5) + (threadIdx.x >> 5);
  int mtiles = (nodes + 15) >> 4;
  if (wave >= mtiles) return;                    // whole-wave exit: EXEC all-1s for WMMA
  int mt = wave;
  int lane = threadIdx.x & 31;
  int kg = lane >> 4;                            // lane half-group
  int m = (mt << 4) + (lane & 15);
  int mload = (m < nodes) ? m : (nodes - 1);     // clamp loads, guard stores
  const float* arow = agg + (long long)mload * 64;
  const float* xrow = x   + (long long)mload * 64;
  float dinv = deginv[mload];
  const v16h* Bl = (const v16h*)WlP;             // [nt][kc][lane] fragments
  const v16h* Br = (const v16h*)WrP;

  v8f acc[NT];
#pragma unroll
  for (int nt = 0; nt < NT; ++nt) acc[nt] = (v8f){};

#pragma unroll
  for (int kc = 0; kc < 2; ++kc) {
    v16h a = load_a_frag(arow, kc * 32 + kg * 8, dinv);
#pragma unroll
    for (int nt = 0; nt < NT; ++nt) {
      v16h b = Bl[(nt * 2 + kc) * 32 + lane];
      acc[nt] = __builtin_amdgcn_wmma_f32_16x16x32_f16(false, a, false, b, (short)0,
                                                       acc[nt], false, false);
    }
  }
#pragma unroll
  for (int kc = 0; kc < 2; ++kc) {
    v16h a = load_a_frag(xrow, kc * 32 + kg * 8, 1.0f);
#pragma unroll
    for (int nt = 0; nt < NT; ++nt) {
      v16h b = Br[(nt * 2 + kc) * 32 + lane];
      acc[nt] = __builtin_amdgcn_wmma_f32_16x16x32_f16(false, a, false, b, (short)0,
                                                       acc[nt], false, false);
    }
  }

#pragma unroll
  for (int nt = 0; nt < NT; ++nt) {
    int n = (nt << 4) + (lane & 15);
    if (n < outc) {
      float bv = bias[n];
#pragma unroll
      for (int j = 0; j < 8; ++j) {              // D layout: VGPR j -> M = j + 8*kg, N = lane&15
        int mr = (mt << 4) + j + (kg << 3);
        if (mr < nodes) out[(long long)mr * outc + n] = acc[nt][j] + bv;
      }
    }
  }
}

// per-channel sum & sumsq (total threads multiple of 64 so channel is fixed per thread)
__global__ void bn_stats_kernel(const float* __restrict__ h, float* __restrict__ stats,
                                long long n64) {
  long long tid = (long long)blockIdx.x * blockDim.x + threadIdx.x;
  long long T = (long long)gridDim.x * blockDim.x;
  int c = (int)(tid & 63);
  float s = 0.0f, s2 = 0.0f;
  for (long long i = tid; i < n64; i += T) {
    float v = h[i];
    s += v; s2 += v * v;
  }
  atomicAdd(&stats[c], s);
  atomicAdd(&stats[64 + c], s2);
}

__global__ void bn_params_kernel(const float* __restrict__ stats,
                                 const float* __restrict__ g, const float* __restrict__ be,
                                 float* __restrict__ params, float invN) {
  int c = threadIdx.x;
  if (c >= 64) return;
  float mu  = stats[c] * invN;
  float var = stats[64 + c] * invN - mu * mu;
  float sc = g[c] * rsqrtf(var + EPS_BN);
  params[c]      = sc;
  params[64 + c] = be[c] - mu * sc;
}

__global__ void bn_relu_kernel(float* __restrict__ h, const float* __restrict__ params,
                               long long n64) {
  long long i = (long long)blockIdx.x * blockDim.x + threadIdx.x;
  long long T = (long long)gridDim.x * blockDim.x;
  for (; i < n64; i += T) {
    int c = (int)(i & 63);
    float v = h[i] * params[c] + params[64 + c];
    h[i] = v > 0.0f ? v : 0.0f;
  }
}

// ---------------- host launcher ----------------

extern "C" void kernel_launch(void* const* d_in, const int* in_sizes, int n_in,
                              void* d_out, int out_size, void* d_ws, size_t ws_size,
                              hipStream_t stream) {
  const float*     x   = (const float*)d_in[0];
  const long long* ei  = (const long long*)d_in[1];   // int64 [2, E]
  const float* Wl0 = (const float*)d_in[2];
  const float* Wr0 = (const float*)d_in[3];
  const float* b0  = (const float*)d_in[4];
  const float* Wl1 = (const float*)d_in[5];
  const float* Wr1 = (const float*)d_in[6];
  const float* b1  = (const float*)d_in[7];
  const float* Wl2 = (const float*)d_in[8];
  const float* Wr2 = (const float*)d_in[9];
  const float* b2  = (const float*)d_in[10];
  const float* g0  = (const float*)d_in[11];
  const float* be0 = (const float*)d_in[12];
  const float* g1  = (const float*)d_in[13];
  const float* be1 = (const float*)d_in[14];

  const int N = in_sizes[0] / 64;
  const int E = in_sizes[1] / 2;
  const long long NC = (long long)N * 64;
  const long long* src = ei;
  const long long* dst = ei + E;

  // workspace layout (256B aligned)
  char* base = (char*)d_ws;
  size_t off = 0;
  auto take = [&](size_t bytes) -> char* {
    char* p = base + off;
    off += (bytes + 255) & ~(size_t)255;
    return p;
  };
  float* agg   = (float*)take(NC * 4);
  float* h0    = (float*)take(NC * 4);
  float* h1    = (float*)take(NC * 4);
  float* deg   = (float*)take((size_t)N * 4);
  float* stats = (float*)take(128 * 4);
  float* bnp   = (float*)take(128 * 4);
  _Float16* WlP0 = (_Float16*)take(4 * 2 * 32 * 16 * 2);
  _Float16* WrP0 = (_Float16*)take(4 * 2 * 32 * 16 * 2);
  _Float16* WlP1 = (_Float16*)take(4 * 2 * 32 * 16 * 2);
  _Float16* WrP1 = (_Float16*)take(4 * 2 * 32 * 16 * 2);
  _Float16* WlP2 = (_Float16*)take(3 * 2 * 32 * 16 * 2);
  _Float16* WrP2 = (_Float16*)take(3 * 2 * 32 * 16 * 2);

  const int ZB = 2048;                       // zero-kernel grid
  const int scatterBlocks = (int)(((long long)E * 64 + 255) / 256);
  const int mtiles = (N + 15) / 16;
  const int gemmGrid = (mtiles + 7) / 8;     // 8 waves per 256-thread block

  // degree once
  zero_f32<<<ZB, 256, 0, stream>>>(deg, N);
  degree_kernel<<<(E + 255) / 256, 256, 0, stream>>>(dst, deg, E);
  deg_inv_kernel<<<(N + 255) / 256, 256, 0, stream>>>(deg, N);

  // pack all weights into WMMA B-fragment layout
  packB_kernel<<<(4 * 1024 + 255) / 256, 256, 0, stream>>>(Wl0, WlP0, 64, 4);
  packB_kernel<<<(4 * 1024 + 255) / 256, 256, 0, stream>>>(Wr0, WrP0, 64, 4);
  packB_kernel<<<(4 * 1024 + 255) / 256, 256, 0, stream>>>(Wl1, WlP1, 64, 4);
  packB_kernel<<<(4 * 1024 + 255) / 256, 256, 0, stream>>>(Wr1, WrP1, 64, 4);
  packB_kernel<<<(3 * 1024 + 255) / 256, 256, 0, stream>>>(Wl2, WlP2, 40, 3);
  packB_kernel<<<(3 * 1024 + 255) / 256, 256, 0, stream>>>(Wr2, WrP2, 40, 3);

  // ---- layer 0 ----
  zero_f32<<<ZB, 256, 0, stream>>>(agg, NC);
  scatter_kernel<<<scatterBlocks, 256, 0, stream>>>(src, dst, x, agg, E);
  sage_gemm_kernel<4><<<gemmGrid, 256, 0, stream>>>(agg, deg, x, WlP0, WrP0, b0, h0, N, 64);
  zero_f32<<<1, 128, 0, stream>>>(stats, 128);
  bn_stats_kernel<<<512, 256, 0, stream>>>(h0, stats, NC);
  bn_params_kernel<<<1, 64, 0, stream>>>(stats, g0, be0, bnp, 1.0f / (float)N);
  bn_relu_kernel<<<ZB, 256, 0, stream>>>(h0, bnp, NC);

  // ---- layer 1 ----
  zero_f32<<<ZB, 256, 0, stream>>>(agg, NC);
  scatter_kernel<<<scatterBlocks, 256, 0, stream>>>(src, dst, h0, agg, E);
  sage_gemm_kernel<4><<<gemmGrid, 256, 0, stream>>>(agg, deg, h0, WlP1, WrP1, b1, h1, N, 64);
  zero_f32<<<1, 128, 0, stream>>>(stats, 128);
  bn_stats_kernel<<<512, 256, 0, stream>>>(h1, stats, NC);
  bn_params_kernel<<<1, 64, 0, stream>>>(stats, g1, be1, bnp, 1.0f / (float)N);
  bn_relu_kernel<<<ZB, 256, 0, stream>>>(h1, bnp, NC);

  // ---- layer 2 (writes d_out: [N, 40] f32) ----
  zero_f32<<<ZB, 256, 0, stream>>>(agg, NC);
  scatter_kernel<<<scatterBlocks, 256, 0, stream>>>(src, dst, h1, agg, E);
  sage_gemm_kernel<3><<<gemmGrid, 256, 0, stream>>>(agg, deg, h1, WlP2, WrP2, b2,
                                                    (float*)d_out, N, 40);
  (void)n_in; (void)out_size; (void)ws_size;
}
